// Model_20942260535790
// MI455X (gfx1250) — matrix operand order repaired and checked by reading, not compiled, not run on hardware
//
#include <hip/hip_runtime.h>
#include <hip/hip_bf16.h>
#include <stdint.h>

// ---------------------------------------------------------------------------
// Problem constants (from reference): V=50257, E=H=128, T=2048, B=64, C=32
// gate width 3H = 384. Token index m = t*B + b (time-major like the scan).
// ---------------------------------------------------------------------------
#define GRU_T 2048
#define GRU_B 64
#define GRU_H 128
#define GRU_G 384           // 3*H
#define NTOK  (GRU_T * GRU_B)   // 131072

typedef __bf16 bf16_t;
typedef __attribute__((ext_vector_type(16))) __bf16 v16bf;
typedef __attribute__((ext_vector_type(8)))  float  v8f;
typedef __attribute__((ext_vector_type(4)))  int    v4i;

// ---------------------------------------------------------------------------
// WMMA helpers (CDNA5: v_wmma_f32_16x16x32_bf16, wave32)
// ---------------------------------------------------------------------------
__device__ __forceinline__ v8f wmma_bf16(v16bf a, v16bf b, v8f c) {
  return __builtin_amdgcn_wmma_f32_16x16x32_bf16(
      /*neg_a=*/false, a, /*neg_b=*/false, b,
      /*c_mod=*/(short)0, c, /*reuse_a=*/false, /*reuse_b=*/false);
}

// A fragment, 16x32 bf16 (row-major source, ldk = row stride in elements).
// ISA layout: lanes 0-15 row M=lane hold K={0..7,16..23}; lanes 16-31 same
// rows hold K={8..15,24..31}.
__device__ __forceinline__ v16bf load_frag_a(const bf16_t* A, int ldk,
                                             int row, int k0, int half) {
  v16bf v;
  const bf16_t* p0 = A + row * ldk + k0 + 8 * half;
  const bf16_t* p1 = p0 + 16;
#pragma unroll
  for (int i = 0; i < 8; ++i) { v[i] = p0[i]; v[8 + i] = p1[i]; }
  return v;
}

// B fragment, 32x16 bf16, where B[k][n] = W[n][k] (W row-major [N][K]).
// lanes 0-15 (col n = lane) hold K={0..7,16..23}, lanes 16-31 hold K={8..15,24..31}.
__device__ __forceinline__ v16bf load_frag_b(const bf16_t* W, int ldk,
                                             int ncol, int k0, int half) {
  v16bf v;
  const bf16_t* p = W + ncol * ldk + k0 + 8 * half;
#pragma unroll
  for (int i = 0; i < 8; ++i) { v[i] = p[i]; v[8 + i] = p[16 + i]; }
  return v;
}

// ---------------------------------------------------------------------------
// Async global->LDS copy (CDNA5 ASYNCcnt path) with sync fallback
// ---------------------------------------------------------------------------
#if defined(__AMDGCN__) && __has_builtin(__builtin_amdgcn_global_load_async_to_lds_b128)
#define GRU_HAS_ASYNC 1
#else
#define GRU_HAS_ASYNC 0
#endif

#if defined(__AMDGCN__) && __has_builtin(__builtin_amdgcn_s_wait_asynccnt)
#define GRU_WAIT_ASYNC(n) __builtin_amdgcn_s_wait_asynccnt(n)
#elif defined(__AMDGCN__)
#define GRU_WAIT_ASYNC(n) asm volatile("s_wait_asynccnt %0" ::"i"(n))
#else
#define GRU_WAIT_ASYNC(n)
#endif

// Copies 16*384 floats (24576 B, contiguous) with 192 threads.
__device__ __forceinline__ void copy_gi_slice(float* dst, const float* src,
                                              int tid) {
#if GRU_HAS_ASYNC
  typedef __attribute__((address_space(1))) v4i* gp_t;
  typedef __attribute__((address_space(3))) v4i* lp_t;
  gp_t g = (gp_t)src;
  lp_t l = (lp_t)dst;
#pragma unroll
  for (int r = 0; r < 8; ++r) {
    const int idx = tid + r * 192;   // 16-byte units; 8*192*16 = 24576 B
    __builtin_amdgcn_global_load_async_to_lds_b128(g + idx, l + idx, 0, 0);
  }
#else
  const float4* s = (const float4*)src;
  float4* d = (float4*)dst;
  for (int i = tid; i < (16 * GRU_G) / 4; i += 192) d[i] = s[i];
#endif
}

// ---------------------------------------------------------------------------
// Kernel 1/3: gi[m][0..383] = Arow(m) . W[g][:] + bias[g]
// Layer 0: Arow(m) = emb[x[(m%B)*T + m/B]]  (gather); layer 1: Arow(m)=h[m].
// Block: 512 thr = 16 waves, M-tile 64 tokens, full N=384.
// wave (wm 0..3, wn 0..3): rows 16*wm, cols 96*wn (6 WMMA tiles).
// ---------------------------------------------------------------------------
__global__ __launch_bounds__(512) void gemm_gates(
    float* __restrict__ gi, const float* __restrict__ Asrc,
    const int* __restrict__ xidx, const float* __restrict__ W,
    const float* __restrict__ bias) {
  __shared__ bf16_t sW[GRU_G * GRU_H];   // 96 KB
  __shared__ bf16_t sA[64 * GRU_H];      // 16 KB
  const int tid = threadIdx.x;
  for (int i = tid; i < GRU_G * GRU_H; i += 512) sW[i] = (bf16_t)W[i];

  const int m0 = blockIdx.x * 64;
  {
    const int row = tid >> 3;        // 0..63
    const int c0 = (tid & 7) * 16;   // 0..112
    const float* src;
    if (xidx) {
      const int m = m0 + row;
      const int t = m >> 6, b = m & 63;               // m = t*64 + b
      src = Asrc + (size_t)xidx[b * GRU_T + t] * GRU_H;
    } else {
      src = Asrc + (size_t)(m0 + row) * GRU_H;
    }
#pragma unroll
    for (int i = 0; i < 16; ++i) sA[row * GRU_H + c0 + i] = (bf16_t)src[c0 + i];
  }
  __syncthreads();

  const int wave = tid >> 5, lane = tid & 31;
  const int half = lane >> 4, l15 = lane & 15;
  const int wm = wave >> 2, wn = wave & 3;

  v8f acc[6] = {};
#pragma unroll
  for (int ks = 0; ks < 4; ++ks) {
    v16bf a = load_frag_a(sA + wm * 16 * GRU_H, GRU_H, l15, ks * 32, half);
#pragma unroll
    for (int t = 0; t < 6; ++t) {
      v16bf b = load_frag_b(sW, GRU_H, wn * 96 + t * 16 + l15, ks * 32, half);
      acc[t] = wmma_bf16(a, b, acc[t]);
    }
  }
#pragma unroll
  for (int t = 0; t < 6; ++t) {
    const int col = wn * 96 + t * 16 + l15;
    const float bv = bias[col];
#pragma unroll
    for (int r = 0; r < 8; ++r) {
      const int row = m0 + wm * 16 + r + 8 * half;   // C layout: M = 8*half + r
      gi[(size_t)row * GRU_G + col] = acc[t][r] + bv;
    }
  }
}

// ---------------------------------------------------------------------------
// Kernel 2/4: GRU recurrence. 4 persistent blocks x 192 thr (6 waves).
// Block owns 16 batch rows; per step: gh = h @ W_hh^T (WMMA), then gates.
// W_hh B-fragments (16 x v16bf = 128 VGPRs) are hoisted into registers for
// the whole time loop; gi slices double-buffered via async-to-LDS.
// ---------------------------------------------------------------------------
__global__ __launch_bounds__(192) void gru_scan(
    float* __restrict__ hout, const float* __restrict__ gi,
    const float* __restrict__ Whh, const float* __restrict__ bhh) {
  __shared__ bf16_t sW[GRU_G * GRU_H];       // 96 KB (only used for preload)
  __shared__ float  sGi[2][16 * GRU_G];      // 48 KB double buffer
  __shared__ bf16_t sH[16 * GRU_H];          // 4 KB  (bf16 h for WMMA A)
  __shared__ float  sHf[16 * GRU_H];         // 8 KB  (fp32 h for z*h term)
  __shared__ float  sGh[16 * GRU_G];         // 24 KB (gh staging)

  const int tid = threadIdx.x;
  for (int i = tid; i < GRU_G * GRU_H; i += 192) sW[i] = (bf16_t)Whh[i];
  for (int i = tid; i < 16 * GRU_H; i += 192) {
    sHf[i] = 0.f;
    sH[i] = (bf16_t)0.f;
  }
  const int bb = blockIdx.x * 16;  // this block's batch rows
  copy_gi_slice(&sGi[0][0], gi + (size_t)bb * GRU_G, tid);  // prefetch t=0

  const int wave = tid >> 5, lane = tid & 31;
  const int half = lane >> 4, l15 = lane & 15;

  __syncthreads();  // sW populated

  // Hoist loop-invariant W_hh fragments + biases into registers.
  v16bf bfrag[4][4];   // [ks][tile]
  float bv[4];
#pragma unroll
  for (int ks = 0; ks < 4; ++ks)
#pragma unroll
    for (int tt = 0; tt < 4; ++tt)
      bfrag[ks][tt] =
          load_frag_b(sW, GRU_H, wave * 64 + tt * 16 + l15, ks * 32, half);
#pragma unroll
  for (int tt = 0; tt < 4; ++tt) bv[tt] = bhh[wave * 64 + tt * 16 + l15];

  for (int t = 0; t < GRU_T; ++t) {
    const int buf = t & 1;
    if (t + 1 < GRU_T) {
      copy_gi_slice(&sGi[buf ^ 1][0],
                    gi + ((size_t)(t + 1) * GRU_B + bb) * GRU_G, tid);
      GRU_WAIT_ASYNC(8);   // 8 newer in flight -> current buffer has landed
    } else {
      GRU_WAIT_ASYNC(0);
    }
    __syncthreads();

    // gh = sH(16x128 bf16) @ W_hh^T -> 16x384, wave owns 64 cols (4 tiles)
    v8f acc[4] = {};
#pragma unroll
    for (int ks = 0; ks < 4; ++ks) {
      v16bf a = load_frag_a(sH, GRU_H, l15, ks * 32, half);
#pragma unroll
      for (int tt = 0; tt < 4; ++tt)
        acc[tt] = wmma_bf16(a, bfrag[ks][tt], acc[tt]);
    }
#pragma unroll
    for (int tt = 0; tt < 4; ++tt) {
      const int col = wave * 64 + tt * 16 + l15;
#pragma unroll
      for (int r = 0; r < 8; ++r)
        sGh[(r + 8 * half) * GRU_G + col] = acc[tt][r] + bv[tt];
    }
    __syncthreads();

    // Gates (torch order r,z,n): h' = (1-z)*tanh(in + r*hn) + z*h
    for (int idx = tid; idx < 16 * GRU_H; idx += 192) {
      const int m = idx >> 7, j = idx & 127;
      const float ir = sGi[buf][m * GRU_G + j];
      const float iz = sGi[buf][m * GRU_G + j + 128];
      const float in_ = sGi[buf][m * GRU_G + j + 256];
      const float hr = sGh[m * GRU_G + j];
      const float hz = sGh[m * GRU_G + j + 128];
      const float hn = sGh[m * GRU_G + j + 256];
      const float r = 1.f / (1.f + __expf(-(ir + hr)));
      const float z = 1.f / (1.f + __expf(-(iz + hz)));
      const float n = tanhf(in_ + r * hn);
      const float h = (1.f - z) * n + z * sHf[idx];
      sHf[idx] = h;
      sH[idx] = (bf16_t)h;
      hout[((size_t)t * GRU_B + bb + m) * GRU_H + j] = h;
    }
    __syncthreads();  // sH stable before next step's WMMA reads
  }
}

// ---------------------------------------------------------------------------
// Kernel 5: out[b*T+t][c] = relu(h[m] . fc_w[c] + fc_b[c]),  m = t*B+b
// Block 256 thr = 8 waves; each wave: 16 tokens x 32 outputs (2 tiles).
// ---------------------------------------------------------------------------
__global__ __launch_bounds__(256) void fc_relu(
    float* __restrict__ out, const float* __restrict__ h,
    const float* __restrict__ fw, const float* __restrict__ fb) {
  __shared__ bf16_t sFW[32 * GRU_H];    // 8 KB
  __shared__ bf16_t sA[128 * GRU_H];    // 32 KB
  const int tid = threadIdx.x;
  for (int i = tid; i < 32 * GRU_H; i += 256) sFW[i] = (bf16_t)fw[i];
  const int m0 = blockIdx.x * 128;
  for (int i = tid; i < 128 * GRU_H; i += 256)
    sA[i] = (bf16_t)h[(size_t)m0 * GRU_H + i];
  __syncthreads();

  const int wave = tid >> 5, lane = tid & 31;
  const int half = lane >> 4, l15 = lane & 15;

  v8f acc[2] = {};
  const bf16_t* Aw = sA + wave * 16 * GRU_H;
#pragma unroll
  for (int ks = 0; ks < 4; ++ks) {
    v16bf a = load_frag_a(Aw, GRU_H, l15, ks * 32, half);
#pragma unroll
    for (int tt = 0; tt < 2; ++tt) {
      v16bf b = load_frag_b(sFW, GRU_H, tt * 16 + l15, ks * 32, half);
      acc[tt] = wmma_bf16(a, b, acc[tt]);
    }
  }
#pragma unroll
  for (int tt = 0; tt < 2; ++tt) {
    const int c = tt * 16 + l15;
    const float bv = fb[c];
#pragma unroll
    for (int r = 0; r < 8; ++r) {
      const int m = m0 + wave * 16 + r + 8 * half;
      const int t = m >> 6, b = m & 63;           // m = t*64 + b
      const float v = acc[tt][r] + bv;
      out[((size_t)b * GRU_T + t) * 32 + c] = v > 0.f ? v : 0.f;
    }
  }
}

// ---------------------------------------------------------------------------
// Host launcher
// ---------------------------------------------------------------------------
extern "C" void kernel_launch(void* const* d_in, const int* in_sizes, int n_in,
                              void* d_out, int out_size, void* d_ws,
                              size_t ws_size, hipStream_t stream) {
  (void)in_sizes; (void)n_in; (void)out_size; (void)ws_size;
  const int*   x     = (const int*)d_in[0];
  const float* emb   = (const float*)d_in[1];
  const float* w_ih0 = (const float*)d_in[2];
  const float* w_hh0 = (const float*)d_in[3];
  const float* b_ih0 = (const float*)d_in[4];
  const float* b_hh0 = (const float*)d_in[5];
  const float* w_ih1 = (const float*)d_in[6];
  const float* w_hh1 = (const float*)d_in[7];
  const float* b_ih1 = (const float*)d_in[8];
  const float* b_hh1 = (const float*)d_in[9];
  const float* fc_w  = (const float*)d_in[10];
  const float* fc_b  = (const float*)d_in[11];
  float* out = (float*)d_out;

  // Workspace: gi [NTOK x 384] f32 (reused by both layers), h [NTOK x 128] f32
  // (layer-1 output aliases layer-0's buffer once layer-1's gi is built).
  float* gi = (float*)d_ws;
  float* h  = (float*)((char*)d_ws + (size_t)NTOK * GRU_G * sizeof(float));

  // Layer 0
  gemm_gates<<<NTOK / 64, 512, 0, stream>>>(gi, emb, x, w_ih0, b_ih0);
  gru_scan<<<4, 192, 0, stream>>>(h, gi, w_hh0, b_hh0);
  // Layer 1
  gemm_gates<<<NTOK / 64, 512, 0, stream>>>(gi, h, nullptr, w_ih1, b_ih1);
  gru_scan<<<4, 192, 0, stream>>>(h, gi, w_hh1, b_hh1);
  // FC + ReLU with [T,B] -> [B,T] permutation folded into stores
  fc_relu<<<NTOK / 128, 256, 0, stream>>>(out, h, fc_w, fc_b);
}
